// Unet_infant_51505247813774
// MI455X (gfx1250) — matrix operand order repaired and checked by reading, not compiled, not run on hardware
//
#include <hip/hip_runtime.h>
#include <hip/hip_bf16.h>

typedef __attribute__((ext_vector_type(16))) _Float16 v16h;
typedef __attribute__((ext_vector_type(8)))  _Float16 v8h;
typedef __attribute__((ext_vector_type(4)))  _Float16 v4h;
typedef __attribute__((ext_vector_type(8)))  float    v8f;

// ---------------------------------------------------------------------------
// FAST fused gather + GEMM using V_WMMA_F32_16X16X32_F16.
// Preconditions: K % 32 == 0, and (neigh != nullptr implies Cin % 32 == 0).
//   Y[M x N] = A[M x K] * W[K x N] + bias
//   A[m,k] = neigh ? X[ neigh[m*7 + k/Cin] * Cin + (k%Cin) ] : X[m*K + k]
// Block = 128 threads (4 waves). Block tile = 32 (M) x 64 (N): each wave owns
// two 16x16 output tiles sharing one B fragment (2 WMMAs / K-step).
// A 32-wide K window never straddles a neighbor segment, so A rows are 32
// contiguous floats -> float4 global loads, packed-f16 b64 LDS stores,
// b128 LDS fragment reads. Segment index tracked by compare+increment
// (no division anywhere in this kernel).
// ---------------------------------------------------------------------------
__global__ __launch_bounds__(128)
void unet_gemm_wmma_fast(const float* __restrict__ X, const int* __restrict__ neigh,
                         const float* __restrict__ W, const float* __restrict__ bias,
                         float* __restrict__ Y, int M, int K, int N, int Cin)
{
    __shared__ __align__(16) _Float16 As[32][32];   // 2 KB: two 16x32 tiles

    const int tid  = threadIdx.x;
    const int wave = tid >> 5;
    const int lane = tid & 31;
    const int hh   = lane >> 4;      // lane half
    const int r    = lane & 15;
    const int m0   = blockIdx.y * 32;
    const int n0   = blockIdx.x * 64 + wave * 16;
    const int ncol = n0 + r;
    const bool nok = (ncol < N);

    // loader mapping: 128 threads = 16 rows x 8 quads; 2 row-halves
    const int lrow = tid >> 3;          // 0..15
    const int lcol = (tid & 7) << 2;    // 0,4,...,28

    v8f acc0 = {};
    v8f acc1 = {};

    int j = 0, ci = 0;                  // k0 = j*Cin + ci  (gather case)
    for (int k0 = 0; k0 < K; k0 += 32) {
        __syncthreads();
        #pragma unroll
        for (int rh = 0; rh < 2; ++rh) {
            const int row = lrow + rh * 16;
            const int gm  = m0 + row;
            float4 v = make_float4(0.f, 0.f, 0.f, 0.f);
            if (gm < M) {
                const float* src = neigh
                    ? X + (long)neigh[gm * 7 + j] * Cin + ci + lcol
                    : X + (long)gm * K + k0 + lcol;
                v = *(const float4*)src;
            }
            v4h h;
            h[0] = (_Float16)v.x; h[1] = (_Float16)v.y;
            h[2] = (_Float16)v.z; h[3] = (_Float16)v.w;
            *(v4h*)&As[row][lcol] = h;                  // ds_store_b64
        }
        __syncthreads();

        // A fragments (16-bit A 16x32 layout): two contiguous 8-f16 runs/lane
        v8h a0lo = *(const v8h*)&As[r][hh * 8];
        v8h a0hi = *(const v8h*)&As[r][16 + hh * 8];
        v8h a1lo = *(const v8h*)&As[16 + r][hh * 8];
        v8h a1hi = *(const v8h*)&As[16 + r][16 + hh * 8];
        v16h a0, a1;
        #pragma unroll
        for (int q = 0; q < 8; ++q) {
            a0[q] = a0lo[q]; a0[8 + q] = a0hi[q];
            a1[q] = a1lo[q]; a1[8 + q] = a1hi[q];
        }

        // B fragment (16-bit B 32x16 layout): lanes 0-15 K=0..7,16..23; 16-31 K=8..15,24..31
        v16h b;
        #pragma unroll
        for (int q = 0; q < 8; ++q) {
            const int kk = k0 + (hh << 3) + (q << 1) + ((q >> 2) << 3);
            float b0 = 0.f, b1 = 0.f;
            if (nok) {
                b0 = W[(long)kk * N + ncol];
                b1 = W[(long)(kk + 1) * N + ncol];
            }
            b[2 * q]     = (_Float16)b0;
            b[2 * q + 1] = (_Float16)b1;
        }

        acc0 = __builtin_amdgcn_wmma_f32_16x16x32_f16(false, a0, false, b, (short)0, acc0, false, false);
        acc1 = __builtin_amdgcn_wmma_f32_16x16x32_f16(false, a1, false, b, (short)0, acc1, false, false);

        ci += 32;                        // exact: Cin % 32 == 0
        if (ci >= Cin) { ci = 0; ++j; }
    }

    if (nok) {
        const float bs = bias ? bias[ncol] : 0.f;
        #pragma unroll
        for (int q = 0; q < 8; ++q) {
            const int row0 = m0 + q + (hh << 3);
            const int row1 = row0 + 16;
            if (row0 < M) Y[(long)row0 * N + ncol] = acc0[q] + bs;
            if (row1 < M) Y[(long)row1 * N + ncol] = acc1[q] + bs;
        }
    }
}

// ---------------------------------------------------------------------------
// GENERIC gather + GEMM (any K, any Cin). Used only for d1-conv1
// (Cin=3, K=21): a single K-step, per-element index math.
// ---------------------------------------------------------------------------
__global__ __launch_bounds__(128)
void unet_gemm_wmma_gen(const float* __restrict__ X, const int* __restrict__ neigh,
                        const float* __restrict__ W, const float* __restrict__ bias,
                        float* __restrict__ Y, int M, int K, int N, int Cin)
{
    __shared__ __align__(16) _Float16 As[32][32];

    const int tid  = threadIdx.x;
    const int wave = tid >> 5;
    const int lane = tid & 31;
    const int hh   = lane >> 4;
    const int r    = lane & 15;
    const int m0   = blockIdx.y * 32;
    const int n0   = blockIdx.x * 64 + wave * 16;
    const int ncol = n0 + r;
    const bool nok = (ncol < N);

    v8f acc0 = {};
    v8f acc1 = {};

    for (int k0 = 0; k0 < K; k0 += 32) {
        __syncthreads();
        for (int e = tid; e < 1024; e += 128) {
            const int row = e >> 5, col = e & 31;
            const int gm = m0 + row, gk = k0 + col;
            float v = 0.f;
            if (gm < M && gk < K) {
                const int j  = gk / Cin;
                const int ci = gk - j * Cin;
                v = X[(long)neigh[gm * 7 + j] * Cin + ci];
            }
            As[row][col] = (_Float16)v;
        }
        __syncthreads();

        v8h a0lo = *(const v8h*)&As[r][hh * 8];
        v8h a0hi = *(const v8h*)&As[r][16 + hh * 8];
        v8h a1lo = *(const v8h*)&As[16 + r][hh * 8];
        v8h a1hi = *(const v8h*)&As[16 + r][16 + hh * 8];
        v16h a0, a1;
        #pragma unroll
        for (int q = 0; q < 8; ++q) {
            a0[q] = a0lo[q]; a0[8 + q] = a0hi[q];
            a1[q] = a1lo[q]; a1[8 + q] = a1hi[q];
        }

        v16h b;
        #pragma unroll
        for (int q = 0; q < 8; ++q) {
            const int kk = k0 + (hh << 3) + (q << 1) + ((q >> 2) << 3);
            float b0 = 0.f, b1 = 0.f;
            if (nok) {
                if (kk     < K) b0 = W[(long)kk * N + ncol];
                if (kk + 1 < K) b1 = W[(long)(kk + 1) * N + ncol];
            }
            b[2 * q]     = (_Float16)b0;
            b[2 * q + 1] = (_Float16)b1;
        }

        acc0 = __builtin_amdgcn_wmma_f32_16x16x32_f16(false, a0, false, b, (short)0, acc0, false, false);
        acc1 = __builtin_amdgcn_wmma_f32_16x16x32_f16(false, a1, false, b, (short)0, acc1, false, false);
    }

    if (nok) {
        const float bs = bias ? bias[ncol] : 0.f;
        #pragma unroll
        for (int q = 0; q < 8; ++q) {
            const int row0 = m0 + q + (hh << 3);
            const int row1 = row0 + 16;
            if (row0 < M) Y[(long)row0 * N + ncol] = acc0[q] + bs;
            if (row1 < M) Y[(long)row1 * N + ncol] = acc1[q] + bs;
        }
    }
}

// ---------------------------------------------------------------------------
// BatchNorm batch stats: 4 columns per block, float4 loads, biased var.
// ---------------------------------------------------------------------------
__global__ __launch_bounds__(256)
void unet_bn_stats4(const float* __restrict__ Y, int M, int C,
                    float* __restrict__ mean, float* __restrict__ rstd)
{
    __shared__ float4 ssum[256];
    __shared__ float4 ssq[256];
    const int c0 = blockIdx.x * 4;
    float4 s = make_float4(0.f, 0.f, 0.f, 0.f);
    float4 q = make_float4(0.f, 0.f, 0.f, 0.f);
    for (int rr = threadIdx.x; rr < M; rr += 256) {
        float4 v = *(const float4*)&Y[(long)rr * C + c0];
        s.x += v.x; s.y += v.y; s.z += v.z; s.w += v.w;
        q.x += v.x * v.x; q.y += v.y * v.y; q.z += v.z * v.z; q.w += v.w * v.w;
    }
    ssum[threadIdx.x] = s; ssq[threadIdx.x] = q;
    __syncthreads();
    for (int st = 128; st > 0; st >>= 1) {
        if (threadIdx.x < st) {
            float4 a = ssum[threadIdx.x], b = ssum[threadIdx.x + st];
            a.x += b.x; a.y += b.y; a.z += b.z; a.w += b.w;
            ssum[threadIdx.x] = a;
            float4 c = ssq[threadIdx.x], d = ssq[threadIdx.x + st];
            c.x += d.x; c.y += d.y; c.z += d.z; c.w += d.w;
            ssq[threadIdx.x] = c;
        }
        __syncthreads();
    }
    if (threadIdx.x < 4) {
        const float m = ((const float*)&ssum[0])[threadIdx.x] / (float)M;
        const float v = ((const float*)&ssq[0])[threadIdx.x] / (float)M - m * m;
        mean[c0 + threadIdx.x] = m;
        rstd[c0 + threadIdx.x] = rsqrtf(v + 1e-5f);
    }
}

// BN apply + LeakyReLU(0.2), in place, float4
__global__ void unet_bn_act4(float* __restrict__ Y, long n4, int C,
                             const float* __restrict__ g, const float* __restrict__ be,
                             const float* __restrict__ mean, const float* __restrict__ rstd)
{
    const long i = (long)blockIdx.x * blockDim.x + threadIdx.x;
    if (i >= n4) return;
    const int c = (int)((i << 2) % C);
    float4 v = ((const float4*)Y)[i];
    float* vp = (float*)&v;
    #pragma unroll
    for (int k = 0; k < 4; ++k) {
        float t = g[c + k] * (vp[k] - mean[c + k]) * rstd[c + k] + be[c + k];
        vp[k] = t > 0.f ? t : 0.2f * t;
    }
    ((float4*)Y)[i] = v;
}

// mean over one-ring (vertex + 6 neighbors), float4 channels
__global__ void unet_pool7(const float* __restrict__ X, const int* __restrict__ neigh,
                           float* __restrict__ O, int num, int C)
{
    const int Cv = C >> 2;
    const long i = (long)blockIdx.x * blockDim.x + threadIdx.x;
    if (i >= (long)num * Cv) return;
    const int v  = (int)(i / Cv);
    const int c4 = (int)(i % Cv) << 2;
    float4 s = make_float4(0.f, 0.f, 0.f, 0.f);
    #pragma unroll
    for (int j = 0; j < 7; ++j) {
        float4 t = *(const float4*)&X[(long)neigh[v * 7 + j] * C + c4];
        s.x += t.x; s.y += t.y; s.z += t.z; s.w += t.w;
    }
    const float sc = 1.f / 7.f;
    s.x *= sc; s.y *= sc; s.z *= sc; s.w *= sc;
    *(float4*)&O[(long)v * C + c4] = s;
}

// upconv merge, float4 channels
__global__ void unet_upmerge(const float* __restrict__ Yup, const int* __restrict__ top,
                             const int* __restrict__ down, float* __restrict__ O,
                             int raw, int np, int C)
{
    const int Cv = C >> 2;
    const long i = (long)blockIdx.x * blockDim.x + threadIdx.x;
    if (i >= (long)(raw + np) * Cv) return;
    const int rr = (int)(i / Cv);
    const int c4 = (int)(i % Cv) << 2;
    float4 o;
    if (rr < raw) {
        o = *(const float4*)&Yup[(long)top[rr] * C + c4];
    } else {
        const int p = rr - raw;
        float4 a = *(const float4*)&Yup[(long)down[2 * p] * C + c4];
        float4 b = *(const float4*)&Yup[(long)down[2 * p + 1] * C + c4];
        o = make_float4(0.5f * (a.x + b.x), 0.5f * (a.y + b.y),
                        0.5f * (a.z + b.z), 0.5f * (a.w + b.w));
    }
    *(float4*)&O[(long)rr * C + c4] = o;
}

// channel concat, float4 (Ca, Cb multiples of 4)
__global__ void unet_concat(const float* __restrict__ A, const float* __restrict__ B,
                            float* __restrict__ O, int M, int Ca, int Cb)
{
    const int Ct = Ca + Cb, Cv = Ct >> 2;
    const long i = (long)blockIdx.x * blockDim.x + threadIdx.x;
    if (i >= (long)M * Cv) return;
    const int rr = (int)(i / Cv);
    const int c4 = (int)(i % Cv) << 2;
    float4 v = (c4 < Ca) ? *(const float4*)&A[(long)rr * Ca + c4]
                         : *(const float4*)&B[(long)rr * Cb + (c4 - Ca)];
    *(float4*)&O[(long)rr * Ct + c4] = v;
}

// ---------------------------------------------------------------------------
extern "C" void kernel_launch(void* const* d_in, const int* in_sizes, int n_in,
                              void* d_out, int out_size, void* d_ws, size_t ws_size,
                              hipStream_t stream)
{
    (void)in_sizes; (void)n_in; (void)out_size; (void)ws_size;

    const float* x        = (const float*)d_in[0];
    const int* n40962     = (const int*)d_in[1];
    const int* n10242     = (const int*)d_in[2];
    const int* n2562      = (const int*)d_in[3];
    const int* n642       = (const int*)d_in[4];
    const int* n162       = (const int*)d_in[5];
    const int* top162     = (const int*)d_in[6];
    const int* down162    = (const int*)d_in[7];
    const int* top642     = (const int*)d_in[8];
    const int* down642    = (const int*)d_in[9];
    const int* top2562    = (const int*)d_in[10];
    const int* down2562   = (const int*)d_in[11];
    const int* top10242   = (const int*)d_in[12];
    const int* down10242  = (const int*)d_in[13];
    auto P = [&](int i) { return (const float*)d_in[14 + i]; };

    char* ws = (char*)d_ws;
    size_t off = 0;
    auto alloc = [&](size_t nf) -> float* {
        float* p = (float*)(ws + off);
        off += ((nf * sizeof(float) + 255) & ~(size_t)255);
        return p;
    };
    float* f0 = alloc(40962UL * 128);
    float* f1 = alloc(40962UL * 128);
    float* f2 = alloc(40962UL * 128);
    float* a2 = alloc(40962UL * 64);
    float* a3 = alloc(10242UL * 128);
    float* a4 = alloc(2562UL * 256);
    float* a5 = alloc(642UL * 512);
    float* a6 = alloc(162UL * 1024);
    float* mean = alloc(1024);
    float* rstd = alloc(1024);

    auto gemm = [&](const float* X, const int* nb, const float* W, const float* bias,
                    float* Y, int M, int K, int N, int Cin) {
        dim3 grid((N + 63) / 64, (M + 31) / 32);
        const bool fast = ((K & 31) == 0) && (nb == nullptr || (Cin & 31) == 0);
        if (fast)
            unet_gemm_wmma_fast<<<grid, 128, 0, stream>>>(X, nb, W, bias, Y, M, K, N, Cin);
        else
            unet_gemm_wmma_gen<<<grid, 128, 0, stream>>>(X, nb, W, bias, Y, M, K, N, Cin);
    };
    auto bn = [&](float* Y, int M, int C, const float* g, const float* be) {
        unet_bn_stats4<<<C / 4, 256, 0, stream>>>(Y, M, C, mean, rstd);
        long n4 = (long)M * C / 4;
        unet_bn_act4<<<(int)((n4 + 255) / 256), 256, 0, stream>>>(Y, n4, C, g, be, mean, rstd);
    };
    auto dc = [&](const float* Xin, const int* nb, int M, int Cin, int Cout,
                  int pb, float* tmp, float* outb) {
        gemm(Xin, nb, P(pb + 0), P(pb + 1), tmp, M, 7 * Cin, Cout, Cin);
        bn(tmp, M, Cout, P(pb + 2), P(pb + 3));
        gemm(tmp, nb, P(pb + 4), P(pb + 5), outb, M, 7 * Cout, Cout, Cout);
        bn(outb, M, Cout, P(pb + 6), P(pb + 7));
    };
    auto pool = [&](const float* X, const int* nb, float* O, int num, int C) {
        long tot = (long)num * C / 4;
        unet_pool7<<<(int)((tot + 255) / 256), 256, 0, stream>>>(X, nb, O, num, C);
    };
    auto upmerge = [&](const float* Yup, const int* top, const int* down, float* O,
                       int raw, int np, int C) {
        long tot = (long)(raw + np) * C / 4;
        unet_upmerge<<<(int)((tot + 255) / 256), 256, 0, stream>>>(Yup, top, down, O, raw, np, C);
    };
    auto concat = [&](const float* A, const float* B, float* O, int M, int Ca, int Cb) {
        long tot = (long)M * (Ca + Cb) / 4;
        unet_concat<<<(int)((tot + 255) / 256), 256, 0, stream>>>(A, B, O, M, Ca, Cb);
    };

    // ---------------- encoder ----------------
    dc(x,  n40962, 40962, 3,    64,   0, f0, a2);          // x2: 40962 x 64
    pool(a2, n40962, f0, 10242, 64);
    dc(f0, n10242, 10242, 64,  128,   8, f1, a3);          // x3: 10242 x 128
    pool(a3, n10242, f0, 2562, 128);
    dc(f0, n2562,  2562, 128,  256,  16, f1, a4);          // x4: 2562 x 256
    pool(a4, n2562, f0, 642, 256);
    dc(f0, n642,    642, 256,  512,  24, f1, a5);          // x5: 642 x 512
    pool(a5, n642, f0, 162, 512);
    dc(f0, n162,    162, 512, 1024,  32, f1, a6);          // x6: 162 x 1024

    // ---------------- decoder ----------------
    // U1: 162 -> 642, cout 512   (params: 40 uw, 41 ub, 42.. dc)
    gemm(a6, nullptr, P(40), P(41), f0, 162, 1024, 3584, 1024);   // (1134 x 512)
    upmerge(f0, top162, down162, f1, 162, 480, 512);              // 642 x 512
    concat(f1, a5, f2, 642, 512, 512);                            // 642 x 1024
    dc(f2, n642, 642, 1024, 512, 42, f0, f1);                     // f1: 642 x 512

    // U2: 642 -> 2562, cout 256  (50 uw, 51 ub, 52.. dc)
    gemm(f1, nullptr, P(50), P(51), f0, 642, 512, 1792, 512);     // (4494 x 256)
    upmerge(f0, top642, down642, f2, 642, 1920, 256);             // 2562 x 256
    concat(f2, a4, f0, 2562, 256, 256);                           // 2562 x 512
    dc(f0, n2562, 2562, 512, 256, 52, f1, f2);                    // f2: 2562 x 256

    // U3: 2562 -> 10242, cout 128 (60 uw, 61 ub, 62.. dc)
    gemm(f2, nullptr, P(60), P(61), f0, 2562, 256, 896, 256);     // (17934 x 128)
    upmerge(f0, top2562, down2562, f1, 2562, 7680, 128);          // 10242 x 128
    concat(f1, a3, f2, 10242, 128, 128);                          // 10242 x 256
    dc(f2, n10242, 10242, 256, 128, 62, f0, f1);                  // f1: 10242 x 128

    // U4: 10242 -> 40962, cout 64 (70 uw, 71 ub, 72.. dc)
    gemm(f1, nullptr, P(70), P(71), f0, 10242, 128, 448, 128);    // (71694 x 64)
    upmerge(f0, top10242, down10242, f2, 10242, 30720, 64);       // 40962 x 64
    concat(f2, a2, f0, 40962, 64, 64);                            // 40962 x 128
    dc(f0, n40962, 40962, 128, 64, 72, f1, f2);                   // f2: 40962 x 64

    // output head: 40962 x 64 @ 64 x 36  (80 ow, 81 ob)
    gemm(f2, nullptr, P(80), P(81), (float*)d_out, 40962, 64, 36, 64);
}
